// TransducerJointNetworkV1_25520695673520
// MI455X (gfx1250) — matrix operand order repaired and checked by reading, not compiled, not run on hardware
//
#include <hip/hip_runtime.h>

#define B_ 4
#define T_ 160
#define S_ 80
#define F_ 640
#define H_ 1024
#define V_ 1024
#define M_ (B_ * T_ * S_) /* 51200 */

#define W1T_BYTES ((size_t)F_ * H_ * 2) /* 1,310,720 */
#define W2T_BYTES ((size_t)H_ * V_ * 2) /* 2,097,152 */
#define WS_NEEDED (W1T_BYTES + W2T_BYTES)

typedef __bf16 v16bf __attribute__((ext_vector_type(16)));
typedef __bf16 v8bf  __attribute__((ext_vector_type(8)));
typedef __bf16 v4bf  __attribute__((ext_vector_type(4)));
typedef float  v8f   __attribute__((ext_vector_type(8)));
typedef float  v4f   __attribute__((ext_vector_type(4)));

static __device__ __forceinline__ v16bf concat8(v8bf lo, v8bf hi) {
  return __builtin_shufflevector(lo, hi, 0, 1, 2, 3, 4, 5, 6, 7,
                                         8, 9, 10, 11, 12, 13, 14, 15);
}

// ---------------------------------------------------------------------------
// Weight transpose+convert: Wt[n][k] = bf16(W[k][n]).  Runs once per launch;
// result (3.3 MB) lives in the 192 MB L2 for the whole GEMM phase.
// ---------------------------------------------------------------------------
__global__ __launch_bounds__(256) void transpose_bf16_kernel(
    const float* __restrict__ W, __bf16* __restrict__ Wt, int K, int N) {
  __shared__ __bf16 tile[32][33];
  const int n0 = blockIdx.x * 32;
  const int k0 = blockIdx.y * 32;
  const int tx = threadIdx.x & 31;
  const int ty = threadIdx.x >> 5;  // 0..7
#pragma unroll
  for (int i = 0; i < 32; i += 8)
    tile[ty + i][tx] = (__bf16)W[(size_t)(k0 + ty + i) * N + n0 + tx];
  __syncthreads();
#pragma unroll
  for (int i = 0; i < 32; i += 8)
    Wt[(size_t)(n0 + ty + i) * K + k0 + tx] = tile[tx][ty + i];
}

// ---------------------------------------------------------------------------
// Kernel 1 (fast path): h = relu((src+tgt) @ W1 + b1) using N-major bf16 W1t.
// Block 256 thr (8 waves), tile 128(M) x 128(N); waves 4(M) x 2(N), each wave
// 2 M-tiles x 4 N-tiles -> 8 wmma / K-step with 4 ds_load_b128 + 8 global
// b128 (1:1 wmma:b128).  On-the-fly A tile staged in 8 KB LDS, pipelined.
// ---------------------------------------------------------------------------
__global__ __launch_bounds__(256) void gemm1_relu_tw_kernel(
    const float* __restrict__ src, const float* __restrict__ tgt,
    const __bf16* __restrict__ W1t, const float* __restrict__ b1,
    float* __restrict__ hOut) {
  __shared__ __align__(64) __bf16 sA[128][32];  // 8 KB

  const int tid = threadIdx.x;
  const int m0  = blockIdx.x * 128;
  const int n0  = blockIdx.y * 128;

  const int w     = tid >> 5;
  const int lane  = tid & 31;
  const int laneN = lane & 15;
  const int hi    = lane >> 4;
  const int wm    = w & 3;   // 2 M-tiles at rows wm*32 (+0, +16)
  const int wn    = w >> 2;  // 64-col group

  // A staging: 2 threads/row, 16 contiguous k each
  const int ar = tid >> 1;        // 0..127
  const int ak = (tid & 1) * 16;  // 0 or 16
  const int m  = m0 + ar;
  const int bi = m / (T_ * S_);
  const int r2 = m % (T_ * S_);
  const int tt = r2 / S_;
  const int ss = r2 % S_;
  const float* srcRow = src + (size_t)(bi * T_ + tt) * F_;
  const float* tgtRow = tgt + (size_t)(bi * S_ + ss) * F_;

  const __bf16* bp[4];
#pragma unroll
  for (int nt = 0; nt < 4; nt++)
    bp[nt] = W1t + (size_t)(n0 + wn * 64 + nt * 16 + laneN) * F_;

  const __bf16* aRow0 = &sA[wm * 32 + laneN][0];
  const __bf16* aRow1 = &sA[wm * 32 + 16 + laneN][0];
  const int kb = hi ? 16 : 0;

  v8f acc[2][4];
#pragma unroll
  for (int mi = 0; mi < 2; mi++)
#pragma unroll
    for (int nt = 0; nt < 4; nt++) acc[mi][nt] = (v8f){0, 0, 0, 0, 0, 0, 0, 0};

  // pipeline prologue: fetch chunk 0 (16 src + 16 tgt floats per thread)
  v4f s[4], t[4];
#pragma unroll
  for (int q = 0; q < 4; q++) {
    s[q] = *(const v4f*)(srcRow + ak + q * 4);
    t[q] = *(const v4f*)(tgtRow + ak + q * 4);
  }

  for (int k0 = 0; k0 < F_; k0 += 32) {
    __syncthreads();  // previous compute done reading sA
    {
      v8bf o0, o1;
#pragma unroll
      for (int i = 0; i < 4; i++) {
        o0[i]     = (__bf16)(s[0][i] + t[0][i]);
        o0[i + 4] = (__bf16)(s[1][i] + t[1][i]);
        o1[i]     = (__bf16)(s[2][i] + t[2][i]);
        o1[i + 4] = (__bf16)(s[3][i] + t[3][i]);
      }
      *(v8bf*)(&sA[ar][ak])     = o0;
      *(v8bf*)(&sA[ar][ak + 8]) = o1;
    }
    __syncthreads();

    if (k0 + 32 < F_) {  // prefetch next chunk while computing
#pragma unroll
      for (int q = 0; q < 4; q++) {
        s[q] = *(const v4f*)(srcRow + k0 + 32 + ak + q * 4);
        t[q] = *(const v4f*)(tgtRow + k0 + 32 + ak + q * 4);
      }
    }

    v16bf af0 = concat8(*(const v8bf*)(aRow0 + kb), *(const v8bf*)(aRow0 + kb + 8));
    v16bf af1 = concat8(*(const v8bf*)(aRow1 + kb), *(const v8bf*)(aRow1 + kb + 8));

    const int krun = k0 + (hi ? 16 : 0);
    v16bf bf[4];
#pragma unroll
    for (int nt = 0; nt < 4; nt++) bf[nt] = *(const v16bf*)(bp[nt] + krun);
#pragma unroll
    for (int nt = 0; nt < 4; nt++) {
      acc[0][nt] = __builtin_amdgcn_wmma_f32_16x16x32_bf16(
          false, af0, false, bf[nt], (short)0, acc[0][nt], false, false);
      acc[1][nt] = __builtin_amdgcn_wmma_f32_16x16x32_bf16(
          false, af1, false, bf[nt], (short)0, acc[1][nt], false, false);
    }
  }

#pragma unroll
  for (int nt = 0; nt < 4; nt++) {
    const int gc = n0 + wn * 64 + nt * 16 + laneN;
    const float bias = b1[gc];
#pragma unroll
    for (int mi = 0; mi < 2; mi++) {
#pragma unroll
      for (int i = 0; i < 8; i++) {
        const int gr = m0 + wm * 32 + mi * 16 + i + hi * 8;
        hOut[(size_t)gr * H_ + gc] = fmaxf(acc[mi][nt][i] + bias, 0.0f);
      }
    }
  }
}

// ---------------------------------------------------------------------------
// Kernel 2 (fast path): logits = h @ W2 + b2 in place, N-major bf16 W2t.
// 32 rows/block (64 KB LDS), 512 threads = 16 waves as 2(M) x 8(N); per
// K-step: 1 A-frag (2 ds_load_b128) + 8 B-frags + 8 back-to-back wmma, zero
// barriers in the main loop.  Halves W2t L2 traffic vs a 16-row tile.
// ---------------------------------------------------------------------------
__global__ __launch_bounds__(512) void gemm2_tw_kernel(
    float* __restrict__ out, const __bf16* __restrict__ W2t,
    const float* __restrict__ b2) {
  __shared__ __align__(64) __bf16 sH[32][H_];  // 64 KB

  const int tid = threadIdx.x;
  const int m0  = blockIdx.x * 32;

  {  // buffer all 32 h rows before any write (in-place safety)
    const int r  = tid >> 4;          // 0..31
    const int cb = (tid & 15) * 64;   // 0..960
    const float* p = out + (size_t)(m0 + r) * H_ + cb;
#pragma unroll
    for (int j = 0; j < 16; j++) {
      v4f v = *(const v4f*)(p + j * 4);
      v4bf o;
#pragma unroll
      for (int e = 0; e < 4; e++) o[e] = (__bf16)v[e];
      *(v4bf*)(&sH[r][cb + j * 4]) = o;
    }
  }
  __syncthreads();

  const int w     = tid >> 5;   // 0..15
  const int lane  = tid & 31;
  const int laneN = lane & 15;
  const int hi    = lane >> 4;
  const int wm    = w & 1;      // M-tile (rows wm*16)
  const int wn    = w >> 1;     // 0..7 -> 128-col group

  const __bf16* bp[8];
#pragma unroll
  for (int j = 0; j < 8; j++)
    bp[j] = W2t + (size_t)(wn * 128 + j * 16 + laneN) * H_;

  const __bf16* aRow = &sH[wm * 16 + laneN][0];

  v8f acc[8];
#pragma unroll
  for (int j = 0; j < 8; j++) acc[j] = (v8f){0, 0, 0, 0, 0, 0, 0, 0};

  for (int k0 = 0; k0 < H_; k0 += 32) {
    const int kb = k0 + (hi ? 16 : 0);
    v16bf afrag = concat8(*(const v8bf*)(aRow + kb), *(const v8bf*)(aRow + kb + 8));

    const int krun = k0 + (hi ? 16 : 0);
#pragma unroll
    for (int j = 0; j < 8; j++) {
      v16bf bfrag = *(const v16bf*)(bp[j] + krun);
      acc[j] = __builtin_amdgcn_wmma_f32_16x16x32_bf16(
          false, afrag, false, bfrag, (short)0, acc[j], false, false);
    }
  }

#pragma unroll
  for (int j = 0; j < 8; j++) {
    const int gc = wn * 128 + j * 16 + laneN;
    const float bias = b2[gc];
#pragma unroll
    for (int i = 0; i < 8; i++) {
      const int gr = m0 + wm * 16 + i + hi * 8;
      out[(size_t)gr * V_ + gc] = acc[j][i] + bias;
    }
  }
}

// ---------------------------------------------------------------------------
// Fallback path (no/undersized workspace): LDS-staged variants.
// ---------------------------------------------------------------------------
__global__ __launch_bounds__(256) void gemm1_relu_fb_kernel(
    const float* __restrict__ src, const float* __restrict__ tgt,
    const float* __restrict__ W1, const float* __restrict__ b1,
    float* __restrict__ hOut) {
  __shared__ __align__(64) __bf16 sA[64][32];
  __shared__ __align__(64) __bf16 sB[128][32];

  const int tid = threadIdx.x;
  const int m0 = blockIdx.x * 64;
  const int n0 = blockIdx.y * 128;
  const int w = tid >> 5, lane = tid & 31, laneN = lane & 15, hi = lane >> 4;
  const int wm = w & 3, wn = w >> 2;
  const int ar = tid >> 2, ak = (tid & 3) * 8;
  const int m = m0 + ar;
  const int bi = m / (T_ * S_), r2 = m % (T_ * S_);
  const int tt = r2 / S_, ss = r2 % S_;
  const float* srcRow = src + (size_t)(bi * T_ + tt) * F_;
  const float* tgtRow = tgt + (size_t)(bi * S_ + ss) * F_;
  const int bk = tid >> 3, bc = (tid & 7) * 16;

  v8f acc[4];
#pragma unroll
  for (int i = 0; i < 4; i++) acc[i] = (v8f){0, 0, 0, 0, 0, 0, 0, 0};

  for (int k0 = 0; k0 < F_; k0 += 32) {
    __syncthreads();
    {
      v4f s0 = *(const v4f*)(srcRow + k0 + ak);
      v4f s1 = *(const v4f*)(srcRow + k0 + ak + 4);
      v4f t0 = *(const v4f*)(tgtRow + k0 + ak);
      v4f t1 = *(const v4f*)(tgtRow + k0 + ak + 4);
      v8bf o;
#pragma unroll
      for (int i = 0; i < 4; i++) {
        o[i] = (__bf16)(s0[i] + t0[i]);
        o[i + 4] = (__bf16)(s1[i] + t1[i]);
      }
      *(v8bf*)(&sA[ar][ak]) = o;
    }
    {
      const float* wrow = W1 + (size_t)(k0 + bk) * H_ + n0 + bc;
#pragma unroll
      for (int j = 0; j < 4; j++) {
        v4f wv = *(const v4f*)(wrow + j * 4);
#pragma unroll
        for (int e = 0; e < 4; e++) sB[bc + j * 4 + e][bk] = (__bf16)wv[e];
      }
    }
    __syncthreads();

    const __bf16* aRow = &sA[wm * 16 + laneN][0];
    const int kb = hi ? 16 : 0;
    v16bf afrag = concat8(*(const v8bf*)(aRow + kb), *(const v8bf*)(aRow + kb + 8));
    const int kr = hi ? 16 : 0;
    v16bf bf[4];
#pragma unroll
    for (int nt = 0; nt < 4; nt++) {
      const __bf16* bRow = &sB[wn * 64 + nt * 16 + laneN][0];
      bf[nt] = concat8(*(const v8bf*)(bRow + kr), *(const v8bf*)(bRow + kr + 8));
    }
#pragma unroll
    for (int nt = 0; nt < 4; nt++)
      acc[nt] = __builtin_amdgcn_wmma_f32_16x16x32_bf16(
          false, afrag, false, bf[nt], (short)0, acc[nt], false, false);
  }

#pragma unroll
  for (int nt = 0; nt < 4; nt++) {
    const int gc = n0 + wn * 64 + nt * 16 + laneN;
    const float bias = b1[gc];
#pragma unroll
    for (int i = 0; i < 8; i++) {
      const int gr = m0 + wm * 16 + i + hi * 8;
      hOut[(size_t)gr * H_ + gc] = fmaxf(acc[nt][i] + bias, 0.0f);
    }
  }
}

__global__ __launch_bounds__(256) void gemm2_fb_kernel(
    float* __restrict__ out, const float* __restrict__ W2,
    const float* __restrict__ b2) {
  __shared__ __align__(64) __bf16 sH[16][H_];
  __shared__ __align__(64) __bf16 sB[128][32];

  const int tid = threadIdx.x;
  const int m0 = blockIdx.x * 16;
  {
    const int r = tid >> 4, cb = (tid & 15) * 64;
    const float* p = out + (size_t)(m0 + r) * H_ + cb;
#pragma unroll
    for (int j = 0; j < 16; j++) {
      v4f v = *(const v4f*)(p + j * 4);
      v4bf o;
#pragma unroll
      for (int e = 0; e < 4; e++) o[e] = (__bf16)v[e];
      *(v4bf*)(&sH[r][cb + j * 4]) = o;
    }
  }
  __syncthreads();

  const int w = tid >> 5, lane = tid & 31, laneN = lane & 15, hi = lane >> 4;
  const int bk = tid >> 3, bc = (tid & 7) * 16;

  for (int n0 = 0; n0 < V_; n0 += 128) {
    v8f acc = (v8f){0, 0, 0, 0, 0, 0, 0, 0};
    for (int k0 = 0; k0 < H_; k0 += 32) {
      __syncthreads();
      {
        const float* wrow = W2 + (size_t)(k0 + bk) * V_ + n0 + bc;
#pragma unroll
        for (int j = 0; j < 4; j++) {
          v4f wv = *(const v4f*)(wrow + j * 4);
#pragma unroll
          for (int e = 0; e < 4; e++) sB[bc + j * 4 + e][bk] = (__bf16)wv[e];
        }
      }
      __syncthreads();
      const __bf16* aRow = &sH[laneN][0];
      const int kb = k0 + (hi ? 16 : 0);
      v16bf afrag = concat8(*(const v8bf*)(aRow + kb), *(const v8bf*)(aRow + kb + 8));
      const __bf16* bRow = &sB[w * 16 + laneN][0];
      const int kr = hi ? 16 : 0;
      v16bf bfrag = concat8(*(const v8bf*)(bRow + kr), *(const v8bf*)(bRow + kr + 8));
      acc = __builtin_amdgcn_wmma_f32_16x16x32_bf16(
          false, afrag, false, bfrag, (short)0, acc, false, false);
    }
    const int gc = n0 + w * 16 + laneN;
    const float bias = b2[gc];
#pragma unroll
    for (int i = 0; i < 8; i++) {
      const int gr = m0 + i + hi * 8;
      out[(size_t)gr * V_ + gc] = acc[i] + bias;
    }
  }
}

// ---------------------------------------------------------------------------
// Kernel 3: in-place log_softmax over V=1024 per row.
// ---------------------------------------------------------------------------
__global__ __launch_bounds__(256) void logsoftmax_kernel(float* __restrict__ out) {
  __shared__ float red[256];
  const int tid = threadIdx.x;
  float* p = out + (size_t)blockIdx.x * V_;

  float x[4];
#pragma unroll
  for (int j = 0; j < 4; j++) x[j] = p[tid + j * 256];

  float lmax = fmaxf(fmaxf(x[0], x[1]), fmaxf(x[2], x[3]));
  red[tid] = lmax;
  __syncthreads();
  for (int s = 128; s > 0; s >>= 1) {
    if (tid < s) red[tid] = fmaxf(red[tid], red[tid + s]);
    __syncthreads();
  }
  const float mmax = red[0];
  __syncthreads();

  float lsum = 0.0f;
#pragma unroll
  for (int j = 0; j < 4; j++) lsum += __expf(x[j] - mmax);
  red[tid] = lsum;
  __syncthreads();
  for (int s = 128; s > 0; s >>= 1) {
    if (tid < s) red[tid] += red[tid + s];
    __syncthreads();
  }
  const float shift = mmax + __logf(red[0]);
#pragma unroll
  for (int j = 0; j < 4; j++) p[tid + j * 256] = x[j] - shift;
}

extern "C" void kernel_launch(void* const* d_in, const int* in_sizes, int n_in,
                              void* d_out, int out_size, void* d_ws,
                              size_t ws_size, hipStream_t stream) {
  (void)in_sizes; (void)n_in; (void)out_size;
  const float* src = (const float*)d_in[0];
  const float* tgt = (const float*)d_in[1];
  const float* W1  = (const float*)d_in[2];
  const float* b1  = (const float*)d_in[3];
  const float* W2  = (const float*)d_in[4];
  const float* b2  = (const float*)d_in[5];
  float* out = (float*)d_out;

  if (d_ws != nullptr && ws_size >= WS_NEEDED) {
    __bf16* W1t = (__bf16*)d_ws;
    __bf16* W2t = (__bf16*)((char*)d_ws + W1T_BYTES);
    transpose_bf16_kernel<<<dim3(H_ / 32, F_ / 32), 256, 0, stream>>>(W1, W1t, F_, H_);
    transpose_bf16_kernel<<<dim3(V_ / 32, H_ / 32), 256, 0, stream>>>(W2, W2t, H_, V_);
    gemm1_relu_tw_kernel<<<dim3(M_ / 128, H_ / 128), 256, 0, stream>>>(src, tgt, W1t, b1, out);
    gemm2_tw_kernel<<<M_ / 32, 512, 0, stream>>>(out, W2t, b2);
  } else {
    gemm1_relu_fb_kernel<<<dim3(M_ / 64, H_ / 128), 256, 0, stream>>>(src, tgt, W1, b1, out);
    gemm2_fb_kernel<<<M_ / 16, 256, 0, stream>>>(out, W2, b2);
  }
  logsoftmax_kernel<<<M_, 256, 0, stream>>>(out);
}
